// EncoderAttention_23983097381434
// MI455X (gfx1250) — matrix operand order repaired and checked by reading, not compiled, not run on hardware
//
#include <hip/hip_runtime.h>

// Problem constants (from reference)
#define NB 16
#define NA 1024
#define CC 512
#define NH 8
#define DD 64
// temperature = sqrt(2*D) = sqrt(128)
#define INV_TP 0.08838834764831845f

typedef __attribute__((ext_vector_type(16))) __bf16 v16bf;
typedef __attribute__((ext_vector_type(8)))  __bf16 v8bf;
typedef __attribute__((ext_vector_type(8)))  float  v8f;

// ---------------------------------------------------------------------------
// Fragment loaders in contiguous k order (k = 16*half + s), same for A and B (not the ISA table order)
// A 16x32 bf16: lane holds row M=(lane&15); half=(lane>>4).
//   a[j]   <- K = 16*half + j       (j=0..7)   contiguous 16B
//   a[8+j] <- K = 16*half + 8 + j   (j=0..7)   contiguous 16B
static __device__ __forceinline__ v16bf load_a_frag(const __bf16* rowk, int hf) {
  v8bf lo = *(const v8bf*)(rowk + 16 * hf);
  v8bf hi = *(const v8bf*)(rowk + 16 * hf + 8);
  v16bf a;
#pragma unroll
  for (int j = 0; j < 8; ++j) { a[j] = lo[j]; a[8 + j] = hi[j]; }
  return a;
}

// B 32x16 bf16: lane holds col N=(lane&15); b[t] <- K = 16*half + t.
// Caller passes pointer already advanced by 16*half along K (contiguous 32B).
static __device__ __forceinline__ v16bf load_b_frag(const __bf16* p) {
  v8bf lo = *(const v8bf*)(p);
  v8bf hi = *(const v8bf*)(p + 8);
  v16bf b;
#pragma unroll
  for (int j = 0; j < 8; ++j) { b[j] = lo[j]; b[8 + j] = hi[j]; }
  return b;
}

// ---------------------------------------------------------------------------
// fp32 -> bf16 conversion (8 elements per thread)
__global__ void __launch_bounds__(256)
cvt_bf16_kernel(const float* __restrict__ src, __bf16* __restrict__ dst, int n8) {
  int i = blockIdx.x * blockDim.x + threadIdx.x;
  if (i >= n8) return;
  const float4* s4 = (const float4*)src;
  float4 a = s4[2 * i];
  float4 b = s4[2 * i + 1];
  v8bf o;
  o[0] = (__bf16)a.x; o[1] = (__bf16)a.y; o[2] = (__bf16)a.z; o[3] = (__bf16)a.w;
  o[4] = (__bf16)b.x; o[5] = (__bf16)b.y; o[6] = (__bf16)b.z; o[7] = (__bf16)b.w;
  *(v8bf*)(dst + (size_t)i * 8) = o;
}

// ---------------------------------------------------------------------------
// QKV projection: out[g, n] = sum_c x[g, c] * W[n, c]   (y = x @ W.T)
// Each wave: 16(M) x 64(N) tile; K loop over CC in steps of 32.
// grid = (16384/128, CC/64(=NH), 3(q,k,v)), block = 256 (8 waves)
// z==0 -> qh row-major, z==1 -> kh row-major, z==2 -> vT[b][h][d][j]
__global__ void __launch_bounds__(256)
qkv_gemm_kernel(const __bf16* __restrict__ xh, const __bf16* __restrict__ wh,
                __bf16* __restrict__ qh, __bf16* __restrict__ kh,
                __bf16* __restrict__ vT) {
  const int wave = threadIdx.x >> 5;
  const int lane = threadIdx.x & 31;
  const int l16 = lane & 15;
  const int hf  = lane >> 4;
  const int g0  = blockIdx.x * 128 + wave * 16;   // output row base (b*NA + i)
  const int hb  = blockIdx.y;                     // 64-wide column block == head
  const int z   = blockIdx.z;                     // 0=q 1=k 2=v
  const __bf16* W = wh + (size_t)z * CC * CC;

  v8f acc[4] = {};
  const __bf16* xrow = xh + (size_t)(g0 + l16) * CC;

  for (int kk = 0; kk < CC; kk += 32) {
    v16bf a = load_a_frag(xrow + kk, hf);
#pragma unroll
    for (int t = 0; t < 4; ++t) {
      int n = hb * 64 + t * 16 + l16;
      v16bf b = load_b_frag(W + (size_t)n * CC + kk + 16 * hf);
      acc[t] = __builtin_amdgcn_wmma_f32_16x16x32_bf16(
          false, a, false, b, (short)0, acc[t], false, false);
    }
  }

  if (z < 2) {
    __bf16* dst = (z == 0) ? qh : kh;
#pragma unroll
    for (int t = 0; t < 4; ++t) {
      int n = hb * 64 + t * 16 + l16;
#pragma unroll
      for (int r = 0; r < 8; ++r) {
        int g = g0 + r + 8 * hf;
        dst[(size_t)g * CC + n] = (__bf16)acc[t][r];
      }
    }
  } else {
    // vT[b][h][d][j]; this tile spans one b (g0 aligned to 16, NA multiple of 128)
    int bb = g0 >> 10;
    int jbase = (g0 & (NA - 1)) + 8 * hf;   // r varies j by +1 -> contiguous v8bf
#pragma unroll
    for (int t = 0; t < 4; ++t) {
      int d = t * 16 + l16;
      v8bf pv;
#pragma unroll
      for (int r = 0; r < 8; ++r) pv[r] = (__bf16)acc[t][r];
      *(v8bf*)(vT + ((size_t)(bb * NH + hb) * DD + d) * NA + jbase) = pv;
    }
  }
}

// ---------------------------------------------------------------------------
// Fused flash attention + bias + mask + residual.
// grid = (NA/128, NH, NB), block = 256 (8 waves); wave = one 16-row query tile.
__global__ void __launch_bounds__(256)
attn_kernel(const __bf16* __restrict__ qh, const __bf16* __restrict__ kh,
            const __bf16* __restrict__ vT, const float* __restrict__ adj,
            const unsigned char* __restrict__ mask, const float* __restrict__ x,
            float* __restrict__ out) {
  __shared__ __bf16 lds_p[8][16][32];   // per-wave P tile staging (8 KB)

  const int wave = threadIdx.x >> 5;
  const int lane = threadIdx.x & 31;
  const int l16  = lane & 15;
  const int hf   = lane >> 4;
  const int b    = blockIdx.z;
  const int h    = blockIdx.y;
  const int i0   = blockIdx.x * 128 + wave * 16;

  // Q fragments for d=0..31 and d=32..63 (K dimension of QK^T)
  const __bf16* qrow = qh + (size_t)(b * NA + i0 + l16) * CC + h * DD;
  v16bf aq0 = load_a_frag(qrow, hf);
  v16bf aq1 = load_a_frag(qrow + 32, hf);

  v8f o[4] = {};
  float mrow[8], lrow[8];
#pragma unroll
  for (int r = 0; r < 8; ++r) { mrow[r] = -1.0e30f; lrow[r] = 0.0f; }

  for (int jb = 0; jb < NA; jb += 32) {
    float s0[8], s1[8];
#pragma unroll
    for (int nt = 0; nt < 2; ++nt) {
      const int j0 = jb + nt * 16;
      const __bf16* krow = kh + (size_t)(b * NA + j0 + l16) * CC + h * DD;
      v16bf bk0 = load_b_frag(krow + 16 * hf);
      v16bf bk1 = load_b_frag(krow + 32 + 16 * hf);
      v8f s = {};
      s = __builtin_amdgcn_wmma_f32_16x16x32_bf16(false, aq0, false, bk0,
                                                  (short)0, s, false, false);
      s = __builtin_amdgcn_wmma_f32_16x16x32_bf16(false, aq1, false, bk1,
                                                  (short)0, s, false, false);
      // scores -> /TP + adjacency, mask -> -1e30 (C-layout: row r+8*hf, col l16)
      size_t base = ((size_t)b * NA + i0 + 8 * hf) * NA + j0 + l16;
      float* sp = nt ? s1 : s0;
#pragma unroll
      for (int r = 0; r < 8; ++r) {
        size_t idx = base + (size_t)r * NA;
        float sv = s[r] * INV_TP + adj[idx];
        if (mask[idx]) sv = -1.0e30f;
        sp[r] = sv;
      }
    }
    // online softmax update per row (reduce over 16 lanes of the half-group)
#pragma unroll
    for (int r = 0; r < 8; ++r) {
      float tm = fmaxf(s0[r], s1[r]);
#pragma unroll
      for (int off = 8; off >= 1; off >>= 1)
        tm = fmaxf(tm, __shfl_xor(tm, off, 32));
      float mn = fmaxf(mrow[r], tm);
      float sc = __expf(mrow[r] - mn);
      mrow[r] = mn;
      float p0 = __expf(s0[r] - mn);
      float p1 = __expf(s1[r] - mn);
      float rs = p0 + p1;
#pragma unroll
      for (int off = 8; off >= 1; off >>= 1)
        rs += __shfl_xor(rs, off, 32);
      lrow[r] = lrow[r] * sc + rs;
      o[0][r] *= sc; o[1][r] *= sc; o[2][r] *= sc; o[3][r] *= sc;
      // stage P (bf16) in this wave's LDS region, C-layout addressing
      lds_p[wave][r + 8 * hf][l16]      = (__bf16)p0;
      lds_p[wave][r + 8 * hf][16 + l16] = (__bf16)p1;
    }
    // wave-local cross-lane exchange: wait for DS stores, then reload in A-layout
    asm volatile("s_wait_dscnt 0" ::: "memory");
    v16bf ap = load_a_frag(&lds_p[wave][l16][0], hf);
    // O += P @ V  (B[K=j][N=d] = vT[b][h][d][j], contiguous along K)
#pragma unroll
    for (int t = 0; t < 4; ++t) {
      const __bf16* vp =
          vT + ((size_t)(b * NH + h) * DD + t * 16 + l16) * NA + jb + 16 * hf;
      v16bf bv = load_b_frag(vp);
      o[t] = __builtin_amdgcn_wmma_f32_16x16x32_bf16(false, ap, false, bv,
                                                     (short)0, o[t], false, false);
    }
    asm volatile("" ::: "memory");  // keep next-iter DS stores after these loads
  }

  // epilogue: normalize, residual add, store fp32
#pragma unroll
  for (int t = 0; t < 4; ++t) {
#pragma unroll
    for (int r = 0; r < 8; ++r) {
      size_t idx = (size_t)(b * NA + i0 + r + 8 * hf) * CC + h * DD + t * 16 + l16;
      out[idx] = o[t][r] / lrow[r] + x[idx];
    }
  }
}

// ---------------------------------------------------------------------------
extern "C" void kernel_launch(void* const* d_in, const int* in_sizes, int n_in,
                              void* d_out, int out_size, void* d_ws, size_t ws_size,
                              hipStream_t stream) {
  const float*         x    = (const float*)d_in[0];
  const unsigned char* mask = (const unsigned char*)d_in[1];
  const float*         adj  = (const float*)d_in[2];
  const float*         Wq   = (const float*)d_in[3];
  const float*         Wk   = (const float*)d_in[4];
  const float*         Wv   = (const float*)d_in[5];
  float*               out  = (float*)d_out;

  // workspace layout (bf16 elements): xh | wh(q,k,v) | qh | kh | vT  ~66 MB
  const size_t XE = (size_t)NB * NA * CC;   // 8,388,608
  const size_t WE = (size_t)CC * CC;        // 262,144
  __bf16* xh = (__bf16*)d_ws;
  __bf16* wh = xh + XE;
  __bf16* qh = wh + 3 * WE;
  __bf16* kh = qh + XE;
  __bf16* vT = kh + XE;

  // 1) convert inputs to bf16
  {
    int n8 = (int)(XE / 8);
    cvt_bf16_kernel<<<(n8 + 255) / 256, 256, 0, stream>>>(x, xh, n8);
    int w8 = (int)(WE / 8);
    cvt_bf16_kernel<<<(w8 + 255) / 256, 256, 0, stream>>>(Wq, wh + 0 * WE, w8);
    cvt_bf16_kernel<<<(w8 + 255) / 256, 256, 0, stream>>>(Wk, wh + 1 * WE, w8);
    cvt_bf16_kernel<<<(w8 + 255) / 256, 256, 0, stream>>>(Wv, wh + 2 * WE, w8);
  }

  // 2) QKV projections (q,k row-major; v transposed to [b,h,d,j])
  qkv_gemm_kernel<<<dim3(NB * NA / 128, CC / 64, 3), 256, 0, stream>>>(
      xh, wh, qh, kh, vT);

  // 3) fused attention + bias + mask + softmax + residual
  attn_kernel<<<dim3(NA / 128, NH, NB), 256, 0, stream>>>(
      qh, kh, vT, adj, mask, x, out);
}
